// InfoNCELoss_10144712753537
// MI455X (gfx1250) — compile-verified, matrix operand order checked by reading, not run on hardware
//
#include <hip/hip_runtime.h>
#include <hip/hip_bf16.h>

typedef _Float16 half4_t __attribute__((ext_vector_type(4)));
typedef _Float16 half8_t __attribute__((ext_vector_type(8)));
typedef _Float16 v16h    __attribute__((ext_vector_type(16)));
typedef float    v8f     __attribute__((ext_vector_type(8)));

#define D_DIM      128
#define TILE_COLS  64
#define LDS_STRIDE 136   // 128 + 8 halves pad: conflict-minimal b128 LDS reads
#define N_SPLITS   4     // column splits for occupancy
#define INV_T      14.285714285714286f   // 1 / 0.07

// ---------------------------------------------------------------------------
// CDNA5 async global->LDS copy (ASYNCcnt-tracked DMA, no VGPR round trip).
// ---------------------------------------------------------------------------
__device__ __forceinline__ void async_copy_b128(unsigned lds_off, const void* gaddr) {
  asm volatile("global_load_async_to_lds_b128 %0, %1, off"
               :: "v"(lds_off), "v"((unsigned long long)(uintptr_t)gaddr)
               : "memory");
}
__device__ __forceinline__ void wait_async_le4() {
  asm volatile("s_wait_asynccnt 0x4" ::: "memory");
}
__device__ __forceinline__ void wait_async_0() {
  asm volatile("s_wait_asynccnt 0x0" ::: "memory");
}

// ---------------------------------------------------------------------------
// Kernel 1: L2-normalize rows of z_i and z_j (fp32 in), emit f16 rows to ws.
// One wave per row; 4 floats per lane; shfl butterfly reduction.
// ---------------------------------------------------------------------------
__global__ __launch_bounds__(256) void infonce_normalize_kernel(
    const float* __restrict__ zi, const float* __restrict__ zj,
    _Float16* __restrict__ zi_h, _Float16* __restrict__ zj_h, int N) {
  int wave = (int)(blockIdx.x * 8u + (threadIdx.x >> 5));
  int lane = threadIdx.x & 31;
  if (wave >= 2 * N) return;

  const float* src;
  _Float16*    dst;
  if (wave < N) { src = zi + (size_t)wave * D_DIM;       dst = zi_h + (size_t)wave * D_DIM; }
  else          { src = zj + (size_t)(wave - N) * D_DIM; dst = zj_h + (size_t)(wave - N) * D_DIM; }

  float4 v = ((const float4*)src)[lane];
  float ss = v.x * v.x + v.y * v.y + v.z * v.z + v.w * v.w;
#pragma unroll
  for (int m = 16; m >= 1; m >>= 1) ss += __shfl_xor(ss, m, 32);
  float r = rsqrtf(ss);

  half4_t h;
  h[0] = (_Float16)(v.x * r);
  h[1] = (_Float16)(v.y * r);
  h[2] = (_Float16)(v.z * r);
  h[3] = (_Float16)(v.w * r);
  *(half4_t*)(dst + lane * 4) = h;
}

// ---------------------------------------------------------------------------
// Kernel 2: fused GEMM + fixed-max online logsumexp, column-split.
// Block = 256 threads = 8 waves; each wave owns a 16-row strip of z_i.
// grid.y = column split; per-row partial (sum, pos) written to ws.
// Double-buffered LDS tiles filled via global_load_async_to_lds_b128.
// ---------------------------------------------------------------------------
__device__ __forceinline__ void issue_tile_async(
    const _Float16* __restrict__ zj_h, int cb, _Float16* tile, int tid) {
#pragma unroll
  for (int i = 0; i < 4; ++i) {
    int f     = tid + 256 * i;    // 0..1023 chunk ids; 64 rows x 16 chunks
    int row   = f >> 4;
    int chunk = f & 15;
    unsigned lds_off = (unsigned)(uintptr_t)(tile + row * LDS_STRIDE + chunk * 8);
    const void* g = zj_h + (size_t)(cb + row) * D_DIM + chunk * 8;
    async_copy_b128(lds_off, g);
  }
}

__global__ __launch_bounds__(256) void infonce_main_kernel(
    const _Float16* __restrict__ zi_h, const _Float16* __restrict__ zj_h,
    float* __restrict__ sumP, float* __restrict__ posP,
    int N, int colsPerSplit) {
  __shared__ __align__(16) _Float16 tiles[2][TILE_COLS * LDS_STRIDE];

  const int tid     = threadIdx.x;
  const int lane    = tid & 31;
  const int waveIdx = tid >> 5;
  const int strip   = blockIdx.x * 8 + waveIdx;
  const int rowBase = strip * 16;
  const int split   = blockIdx.y;
  const int colStart = split * colsPerSplit;

  const int m  = lane & 15;
  const int hi = lane >> 4;

  // Resident A fragments: row (rowBase+m), 4 K-chunks of 32.
  const _Float16* arow = zi_h + (size_t)(rowBase + m) * D_DIM;
  v16h A[4];
#pragma unroll
  for (int kc = 0; kc < 4; ++kc) {
    const int kb = kc * 32;
    half8_t lo = *(const half8_t*)(arow + kb + 8 * hi);
    half8_t hp = *(const half8_t*)(arow + kb + 16 + 8 * hi);
    A[kc] = __builtin_shufflevector(lo, hp, 0, 1, 2, 3, 4, 5, 6, 7,
                                            8, 9, 10, 11, 12, 13, 14, 15);
  }

  float sums[8];
  float pos[8];
#pragma unroll
  for (int r = 0; r < 8; ++r) { sums[r] = 0.0f; pos[r] = 0.0f; }

  const int T = colsPerSplit / TILE_COLS;
  issue_tile_async(zj_h, colStart, tiles[0], tid);

  for (int it = 0; it < T; ++it) {
    const _Float16* cur = tiles[it & 1];
    if (it + 1 < T) {
      __syncthreads();  // all waves done reading the buffer we are about to refill
      issue_tile_async(zj_h, colStart + (it + 1) * TILE_COLS, tiles[(it + 1) & 1], tid);
      wait_async_le4(); // tile `it` (4 older copies) complete; tile it+1 in flight
    } else {
      wait_async_0();
    }
    __syncthreads();    // tile `it` visible from every wave's copies

#pragma unroll
    for (int sub = 0; sub < 4; ++sub) {
      const int colBase = colStart + it * TILE_COLS + sub * 16;
      const _Float16* brow = cur + (sub * 16 + m) * LDS_STRIDE;

      // Load all four B fragments, then run the 4 WMMAs back-to-back.
      v16h Bf[4];
#pragma unroll
      for (int kc = 0; kc < 4; ++kc) {
        const int kb = kc * 32;
        half8_t lo = *(const half8_t*)(brow + kb + 8 * hi);
        half8_t hp = *(const half8_t*)(brow + kb + 16 + 8 * hi);
        Bf[kc] = __builtin_shufflevector(lo, hp, 0, 1, 2, 3, 4, 5, 6, 7,
                                                 8, 9, 10, 11, 12, 13, 14, 15);
      }
      v8f c = {};
#pragma unroll
      for (int kc = 0; kc < 4; ++kc)
        c = __builtin_amdgcn_wmma_f32_16x16x32_f16(
                false, A[kc], false, Bf[kc], (short)0, c, false, false);

#pragma unroll
      for (int r = 0; r < 8; ++r)
        sums[r] += __expf((c[r] - 1.0f) * INV_T);   // fixed-max online LSE

      if (colBase == rowBase) {   // wave-uniform: 1 of 256 tiles -> s_cbranch
#pragma unroll
        for (int r = 0; r < 8; ++r)
          if (m == r + 8 * hi) pos[r] = c[r] * INV_T;
      }
    }
  }

  // Reduce across the 16 lanes sharing each row (xor butterfly stays in group).
#pragma unroll
  for (int msk = 8; msk >= 1; msk >>= 1) {
#pragma unroll
    for (int r = 0; r < 8; ++r) {
      sums[r] += __shfl_xor(sums[r], msk, 32);
      pos[r]  += __shfl_xor(pos[r],  msk, 32);
    }
  }

  // lanes 0..7 (hi=0) hold rows 0..7; lanes 16..23 (hi=1) hold rows 8..15.
  if (m < 8) {
    int row = rowBase + 8 * hi + m;
    sumP[(size_t)split * N + row] = sums[m];
    posP[(size_t)split * N + row] = pos[m];
  }
}

// ---------------------------------------------------------------------------
// Kernel 3: deterministic reduction: combine splits, log, mean.
// ---------------------------------------------------------------------------
__global__ __launch_bounds__(256) void infonce_reduce_kernel(
    const float* __restrict__ sumP, const float* __restrict__ posP,
    float* __restrict__ out, int N) {
  __shared__ float sh[256];
  float acc = 0.0f;
  for (int i = threadIdx.x; i < N; i += 256) {
    float s = 0.0f, p = 0.0f;
#pragma unroll
    for (int k = 0; k < N_SPLITS; ++k) {
      s += sumP[(size_t)k * N + i];
      p += posP[(size_t)k * N + i];
    }
    acc += INV_T + __logf(s) - p;
  }
  sh[threadIdx.x] = acc;
  __syncthreads();
#pragma unroll
  for (int step = 128; step >= 1; step >>= 1) {
    if ((int)threadIdx.x < step) sh[threadIdx.x] += sh[threadIdx.x + step];
    __syncthreads();
  }
  if (threadIdx.x == 0) out[0] = sh[0] / (float)N;
}

extern "C" void kernel_launch(void* const* d_in, const int* in_sizes, int n_in,
                              void* d_out, int out_size, void* d_ws, size_t ws_size,
                              hipStream_t stream) {
  const float* zi = (const float*)d_in[0];
  const float* zj = (const float*)d_in[1];
  const int N = in_sizes[0] / D_DIM;   // 16384

  // Workspace: [zi_h f16 N*128][zj_h f16 N*128][sumP f32 S*N][posP f32 S*N]
  _Float16* zi_h = (_Float16*)d_ws;
  _Float16* zj_h = zi_h + (size_t)N * D_DIM;
  float* sumP = (float*)(zj_h + (size_t)N * D_DIM);
  float* posP = sumP + (size_t)N_SPLITS * N;

  int normBlocks = (2 * N + 7) / 8;
  infonce_normalize_kernel<<<normBlocks, 256, 0, stream>>>(zi, zj, zi_h, zj_h, N);

  dim3 grid(N / 128, N_SPLITS);
  infonce_main_kernel<<<grid, 256, 0, stream>>>(zi_h, zj_h, sumP, posP,
                                                N, N / N_SPLITS);

  infonce_reduce_kernel<<<1, 256, 0, stream>>>(sumP, posP, (float*)d_out, N);
}
